// MoELayer_10402410791603
// MI455X (gfx1250) — matrix operand order repaired and checked by reading, not compiled, not run on hardware
//
#include <hip/hip_runtime.h>
#include <hip/hip_bf16.h>
#include <cstddef>
#include <cstdint>

// ---------------- problem constants ----------------
#define HDIM 1024
#define FDIM 4096
#define ECNT 8
#define NTOK 4096                  // B*T tokens
#define CAP  512                   // NTOK / ECNT
#define NGRP 9                     // 8 experts + shared fallback

// ---------------- GEMM tiling ----------------
#define BM 128
#define BN 128
#define BK 32
#define LDT 56                     // LDS row stride (bf16): 112B = 16B-aligned, bank-friendly

typedef __attribute__((ext_vector_type(16))) __bf16 v16bf;
typedef __attribute__((ext_vector_type(8)))  __bf16 v8bf;
typedef __attribute__((ext_vector_type(4)))  __bf16 v4bf;
typedef __attribute__((ext_vector_type(8)))  float  v8f;

// ---------------- CDNA5 async global->LDS copy (ASYNCcnt path) ----------------
// Builtin signature (from toolchain diagnostic): pointers to 16B int vectors,
// src in addrspace(1), dst in addrspace(3), plus imm offset + imm cpol.
typedef int int4v __attribute__((vector_size(16)));
typedef __attribute__((address_space(1))) int4v as1_int4;
typedef __attribute__((address_space(3))) int4v as3_int4;

__device__ __forceinline__ as1_int4* as_global4(const void* p) {
    return (as1_int4*)(unsigned long long)(uintptr_t)p;
}
__device__ __forceinline__ as3_int4* as_lds4(const void* p) {
    // generic LDS pointer: low 32 bits are the LDS byte offset (aperture in high bits)
    return (as3_int4*)(unsigned)(uintptr_t)p;
}

#if defined(__HIP_DEVICE_COMPILE__)
# if __has_builtin(__builtin_amdgcn_global_load_async_to_lds_b128)
#  define ASYNC_CP16(dst, src) \
      __builtin_amdgcn_global_load_async_to_lds_b128(as_global4(src), as_lds4(dst), 0, 0)
# else
#  define ASYNC_CP16(dst, src) \
      asm volatile("global_load_async_to_lds_b128 %0, %1, off" \
                   :: "v"((unsigned)(uintptr_t)(dst)), \
                      "v"((unsigned long long)(uintptr_t)(src)) : "memory")
# endif
# if __has_builtin(__builtin_amdgcn_s_wait_asynccnt)
#  define WAIT_ASYNC(n) __builtin_amdgcn_s_wait_asynccnt(n)
# else
#  define WAIT_ASYNC(n) asm volatile("s_wait_asynccnt %0" :: "n"(n) : "memory")
# endif
#else
# define ASYNC_CP16(dst, src) ((void)0)
# define WAIT_ASYNC(n)        ((void)0)
#endif

// =====================================================================
// Kernel 1: router. One wave per token: logits = x @ rw^T + rb, top-2.
// =====================================================================
__global__ __launch_bounds__(256)
void router_kernel(const float* __restrict__ xf, const float* __restrict__ rw,
                   const float* __restrict__ rb,
                   int* __restrict__ primary, int* __restrict__ second) {
    const int wave = threadIdx.x >> 5;
    const int lane = threadIdx.x & 31;
    const int n = blockIdx.x * 8 + wave;
    if (n >= NTOK) return;

    float acc[ECNT];
#pragma unroll
    for (int e = 0; e < ECNT; ++e) acc[e] = 0.0f;

    const float* xr = xf + (size_t)n * HDIM;
    for (int k = lane; k < HDIM; k += 32) {
        const float xv = xr[k];
#pragma unroll
        for (int e = 0; e < ECNT; ++e) acc[e] += xv * rw[e * HDIM + k];
    }
#pragma unroll
    for (int e = 0; e < ECNT; ++e) {
#pragma unroll
        for (int off = 16; off > 0; off >>= 1)
            acc[e] += __shfl_xor(acc[e], off, 32);
    }
    if (lane == 0) {
        float l[ECNT];
#pragma unroll
        for (int e = 0; e < ECNT; ++e) l[e] = acc[e] + rb[e];
        int i0 = 0;
#pragma unroll
        for (int e = 1; e < ECNT; ++e) if (l[e] > l[i0]) i0 = e;
        int i1 = (i0 == 0) ? 1 : 0;
#pragma unroll
        for (int e = 0; e < ECNT; ++e) if (e != i0 && l[e] > l[i1]) i1 = e;
        primary[n] = i0;
        second[n]  = i1;
    }
}

// =====================================================================
// Kernel 2: capacity assignment (greedy in-order == reference cumsum logic).
// One wave; lane e owns expert e; lane 0 builds the fallback list.
// =====================================================================
__global__ void assign_kernel(const int* __restrict__ primary, const int* __restrict__ second,
                              int* __restrict__ assign, int* __restrict__ lists,
                              int* __restrict__ counts) {
    const int lane = threadIdx.x;
    int cnt = 0;
    if (lane < ECNT) {
        for (int n = 0; n < NTOK; ++n) {
            if (primary[n] == lane) {
                if (cnt < CAP) { assign[n] = lane; lists[lane * NTOK + cnt] = n; ++cnt; }
                else           { assign[n] = -1; }
            }
        }
    }
    __syncthreads();
    if (lane < ECNT) {
        for (int n = 0; n < NTOK; ++n) {
            if (second[n] == lane && assign[n] == -1 && cnt < CAP) {
                assign[n] = lane; lists[lane * NTOK + cnt] = n; ++cnt;
            }
        }
        counts[lane] = cnt;
    }
    __syncthreads();
    if (lane == 0) {
        int f = 0;
        for (int n = 0; n < NTOK; ++n)
            if (assign[n] == -1) lists[ECNT * NTOK + f++] = n;
        counts[ECNT] = f;
    }
}

// =====================================================================
// WMMA helpers. 128x128x32 tile, 8 waves, each wave: 2x4 of 16x16 tiles.
// =====================================================================
__device__ __forceinline__ v16bf load_a_frag(const __bf16* As, int row, int hi) {
    const __bf16* base = As + row * LDT;
    v8bf lo = *(const v8bf*)(base + hi * 8);
    v8bf hh = *(const v8bf*)(base + 16 + hi * 8);
    return __builtin_shufflevector(lo, hh, 0,1,2,3,4,5,6,7,8,9,10,11,12,13,14,15);
}
__device__ __forceinline__ v16bf load_b_frag(const __bf16* Bs, int col, int hi) {
    const __bf16* base = Bs + col * LDT + hi * 8;
    v8bf lo = *(const v8bf*)(base);
    v8bf hh = *(const v8bf*)(base + 16);
    return __builtin_shufflevector(lo, hh, 0,1,2,3,4,5,6,7,8,9,10,11,12,13,14,15);
}

// B micro-tile: 4(k) x 4(n) per thread; coalesced b128 loads, packed b64 LDS stores.
__device__ __forceinline__ void load_b_regs(const float* W, int ldw, int krow, int ncol,
                                            float4* R) {
    const float* bb = W + (size_t)krow * ldw + ncol;
    R[0] = *(const float4*)(bb);
    R[1] = *(const float4*)(bb + ldw);
    R[2] = *(const float4*)(bb + 2 * ldw);
    R[3] = *(const float4*)(bb + 3 * ldw);
}
__device__ __forceinline__ void store_b_tile(__bf16* Bs, int ng, int kg, const float4* R) {
#pragma unroll
    for (int n = 0; n < 4; ++n) {
        v4bf v;
#pragma unroll
        for (int q = 0; q < 4; ++q) v[q] = (__bf16)((const float*)&R[q])[n];
        *(v4bf*)&Bs[(ng + n) * LDT + kg] = v;   // transposed: Bs[n][k], 8B store
    }
}

// =====================================================================
// Kernel 3: layer-1 FFN.  h = gelu(gather(x) @ W1 + b1) -> bf16 scratch.
// Register-pipelined: next K-tile global loads overlap current WMMA.
// =====================================================================
__global__ __launch_bounds__(256)
void expert_ffn1(const float* __restrict__ xf,
                 const float* __restrict__ w1, const float* __restrict__ b1,
                 const float* __restrict__ sw1, const float* __restrict__ sb1,
                 const int* __restrict__ lists, const int* __restrict__ counts,
                 __bf16* __restrict__ hbuf) {
    const int z   = blockIdx.z;
    const int cnt = counts[z];
    const int mt  = blockIdx.y;
    if (mt * BM >= cnt) return;
    const int n0 = blockIdx.x * BN;

    const float* W    = (z < ECNT) ? (w1 + (size_t)z * HDIM * FDIM) : sw1;
    const float* bias = (z < ECNT) ? (b1 + (size_t)z * FDIM)        : sb1;
    const size_t slotBase = (z < ECNT) ? (size_t)z * CAP : (size_t)ECNT * CAP;

    __shared__ __align__(16) __bf16 As[BM * LDT];
    __shared__ __align__(16) __bf16 Bs[BN * LDT];
    __shared__ int tok[BM];

    const int tid = threadIdx.x;
    if (tid < BM) {
        const int r = mt * BM + tid;
        tok[tid] = (r < cnt) ? lists[z * NTOK + r] : -1;
    }
    __syncthreads();

    const int lane = tid & 31, wave = tid >> 5;
    const int mOff = (wave >> 1) * 32, nOff = (wave & 1) * 64;
    const int lr = lane & 15, hi = lane >> 4;

    const v8f zero8 = {0.f,0.f,0.f,0.f,0.f,0.f,0.f,0.f};
    v8f acc[2][4];
#pragma unroll
    for (int r = 0; r < 2; ++r)
#pragma unroll
        for (int c = 0; c < 4; ++c) acc[r][c] = zero8;

    const int ar = tid >> 1, ac = (tid & 1) * 16;       // A: 2 thr/row, 16 f32
    const int at = tok[ar];
    const float* arow = (at >= 0) ? (xf + (size_t)at * HDIM) : nullptr;
    const int kg = (tid >> 5) * 4;                       // B: 4x4 micro-tile
    const int ng = (tid & 31) * 4;

    float4 aR[4], bR[4];
    const float4 fz = {0.f, 0.f, 0.f, 0.f};
    // prologue: K-tile 0 into registers
    if (arow) {
        const float4* p = (const float4*)(arow + ac);
        aR[0] = p[0]; aR[1] = p[1]; aR[2] = p[2]; aR[3] = p[3];
    } else { aR[0] = fz; aR[1] = fz; aR[2] = fz; aR[3] = fz; }
    load_b_regs(W, FDIM, kg, n0 + ng, bR);

    for (int k0 = 0; k0 < HDIM; k0 += BK) {
        // convert + store current tile to LDS (packed b128/b64 stores)
        {
            v8bf v0, v1;
            const float* a0 = (const float*)&aR[0]; const float* a1 = (const float*)&aR[1];
            const float* a2 = (const float*)&aR[2]; const float* a3 = (const float*)&aR[3];
#pragma unroll
            for (int i = 0; i < 4; ++i) {
                v0[i] = (__bf16)a0[i]; v0[4+i] = (__bf16)a1[i];
                v1[i] = (__bf16)a2[i]; v1[4+i] = (__bf16)a3[i];
            }
            *(v8bf*)&As[ar * LDT + ac]     = v0;
            *(v8bf*)&As[ar * LDT + ac + 8] = v1;
            store_b_tile(Bs, ng, kg, bR);
        }
        __syncthreads();

        // prefetch next K-tile into registers (overlaps WMMA below)
        const int kn = k0 + BK;
        if (kn < HDIM) {
            if (arow) {
                const float4* p = (const float4*)(arow + kn + ac);
                aR[0] = p[0]; aR[1] = p[1]; aR[2] = p[2]; aR[3] = p[3];
            }
            load_b_regs(W, FDIM, kn + kg, n0 + ng, bR);
        }

        v16bf a[2], b[4];
#pragma unroll
        for (int r = 0; r < 2; ++r) a[r] = load_a_frag(As, mOff + r * 16 + lr, hi);
#pragma unroll
        for (int c = 0; c < 4; ++c) b[c] = load_b_frag(Bs, nOff + c * 16 + lr, hi);
#pragma unroll
        for (int r = 0; r < 2; ++r)
#pragma unroll
            for (int c = 0; c < 4; ++c)
                acc[r][c] = __builtin_amdgcn_wmma_f32_16x16x32_bf16(
                    false, a[r], false, b[c], (short)0, acc[r][c], false, false);
        __syncthreads();
    }

    // epilogue: bias + exact gelu -> bf16
#pragma unroll
    for (int r = 0; r < 2; ++r) {
#pragma unroll
        for (int c = 0; c < 4; ++c) {
            const int mBase = mOff + r * 16 + hi * 8;
            const int n = n0 + nOff + c * 16 + lr;
            const float bv = bias[n];
#pragma unroll
            for (int j = 0; j < 8; ++j) {
                const int grow = mt * BM + mBase + j;
                if (grow < cnt) {
                    const float v = acc[r][c][j] + bv;
                    const float g = 0.5f * v * (1.0f + erff(v * 0.70710678118654752f));
                    hbuf[(slotBase + grow) * (size_t)FDIM + n] = (__bf16)g;
                }
            }
        }
    }
}

// =====================================================================
// Kernel 4: layer-2 FFN.  y[token] = h @ W2 + b2, scattered per token list.
// A operand is bf16 already -> double-buffered GLOBAL_LOAD_ASYNC_TO_LDS_B128
// (ASYNCcnt, in-order completion: wait<=2 after issuing next tile's 2 ops).
// B operand register-pipelined as in kernel 3.
// =====================================================================
__global__ __launch_bounds__(256)
void expert_ffn2(const __bf16* __restrict__ hbuf,
                 const float* __restrict__ w2, const float* __restrict__ b2,
                 const float* __restrict__ sw2, const float* __restrict__ sb2,
                 const int* __restrict__ lists, const int* __restrict__ counts,
                 float* __restrict__ out) {
    const int z   = blockIdx.z;
    const int cnt = counts[z];
    const int mt  = blockIdx.y;
    if (mt * BM >= cnt) return;
    const int n0 = blockIdx.x * BN;

    const float* W    = (z < ECNT) ? (w2 + (size_t)z * FDIM * HDIM) : sw2;
    const float* bias = (z < ECNT) ? (b2 + (size_t)z * HDIM)        : sb2;
    const size_t slotBase = (z < ECNT) ? (size_t)z * CAP : (size_t)ECNT * CAP;

    __shared__ __align__(16) __bf16 As[2][BM * LDT];
    __shared__ __align__(16) __bf16 Bs[BN * LDT];
    __shared__ int tok[BM];

    const int tid = threadIdx.x;
    if (tid < BM) {
        const int r = mt * BM + tid;
        tok[tid] = (r < cnt) ? lists[z * NTOK + r] : -1;
    }
    __syncthreads();

    const int lane = tid & 31, wave = tid >> 5;
    const int mOff = (wave >> 1) * 32, nOff = (wave & 1) * 64;
    const int lr = lane & 15, hi = lane >> 4;

    const v8f zero8 = {0.f,0.f,0.f,0.f,0.f,0.f,0.f,0.f};
    v8f acc[2][4];
#pragma unroll
    for (int r = 0; r < 2; ++r)
#pragma unroll
        for (int c = 0; c < 4; ++c) acc[r][c] = zero8;

    const int ar = tid >> 1, ac = (tid & 1) * 16;
    const __bf16* arow = hbuf + (slotBase + mt * BM + ar) * (size_t)FDIM + ac;
    const int kg = (tid >> 5) * 4;
    const int ng = (tid & 31) * 4;

    // prologue: async A tile 0 straight into LDS; B tile 0 into registers
    ASYNC_CP16(&As[0][ar * LDT + ac],     arow);
    ASYNC_CP16(&As[0][ar * LDT + ac + 8], arow + 8);
    float4 bR[4];
    load_b_regs(W, HDIM, kg, n0 + ng, bR);

    const int NIT = FDIM / BK;           // 128
    for (int it = 0; it < NIT; ++it) {
        const int k0 = it * BK;
        const __bf16* Ac = &As[it & 1][0];

        store_b_tile(Bs, ng, kg, bR);

        if (it + 1 < NIT) {
            // next A tile into the other buffer (its readers finished at the
            // end-barrier of iteration it-1), then wait for current tile only.
            __bf16* An = &As[(it + 1) & 1][0];
            ASYNC_CP16(&An[ar * LDT + ac],     arow + k0 + BK);
            ASYNC_CP16(&An[ar * LDT + ac + 8], arow + k0 + BK + 8);
            WAIT_ASYNC(2);
        } else {
            WAIT_ASYNC(0);
        }
        __syncthreads();

        if (it + 1 < NIT) load_b_regs(W, HDIM, k0 + BK + kg, n0 + ng, bR);

        v16bf a[2], b[4];
#pragma unroll
        for (int r = 0; r < 2; ++r) a[r] = load_a_frag(Ac, mOff + r * 16 + lr, hi);
#pragma unroll
        for (int c = 0; c < 4; ++c) b[c] = load_b_frag(Bs, nOff + c * 16 + lr, hi);
#pragma unroll
        for (int r = 0; r < 2; ++r)
#pragma unroll
            for (int c = 0; c < 4; ++c)
                acc[r][c] = __builtin_amdgcn_wmma_f32_16x16x32_bf16(
                    false, a[r], false, b[c], (short)0, acc[r][c], false, false);
        __syncthreads();
    }

    // epilogue: bias, scatter fp32 rows to assigned tokens
#pragma unroll
    for (int r = 0; r < 2; ++r) {
#pragma unroll
        for (int c = 0; c < 4; ++c) {
            const int mBase = mOff + r * 16 + hi * 8;
            const int n = n0 + nOff + c * 16 + lr;
            const float bv = bias[n];
#pragma unroll
            for (int j = 0; j < 8; ++j) {
                const int grow = mt * BM + mBase + j;
                if (grow < cnt) {
                    const int t = tok[mBase + j];
                    out[(size_t)t * HDIM + n] = acc[r][c][j] + bv;
                }
            }
        }
    }
}

// =====================================================================
// Host launcher
// =====================================================================
extern "C" void kernel_launch(void* const* d_in, const int* in_sizes, int n_in,
                              void* d_out, int out_size, void* d_ws, size_t ws_size,
                              hipStream_t stream) {
    const float* x   = (const float*)d_in[0];
    const float* rw  = (const float*)d_in[1];
    const float* rb  = (const float*)d_in[2];
    const float* w1  = (const float*)d_in[3];
    const float* b1  = (const float*)d_in[4];
    const float* w2  = (const float*)d_in[5];
    const float* b2  = (const float*)d_in[6];
    const float* sw1 = (const float*)d_in[7];
    const float* sb1 = (const float*)d_in[8];
    const float* sw2 = (const float*)d_in[9];
    const float* sb2 = (const float*)d_in[10];
    float* out = (float*)d_out;

    char* ws = (char*)d_ws;
    int* primary = (int*)(ws);
    int* second  = (int*)(ws + (size_t)NTOK * 4);
    int* assignb = (int*)(ws + (size_t)NTOK * 8);
    int* counts  = (int*)(ws + (size_t)NTOK * 12);
    int* lists   = (int*)(ws + (size_t)NTOK * 12 + 256);
    // bf16 intermediate: 8 expert slots of CAP rows + fallback slot of NTOK rows
    __bf16* hbuf = (__bf16*)(ws + (size_t)NTOK * 12 + 256 + (size_t)NGRP * NTOK * 4);

    router_kernel<<<dim3(NTOK / 8), 256, 0, stream>>>(x, rw, rb, primary, second);
    assign_kernel<<<dim3(1), 32, 0, stream>>>(primary, second, assignb, lists, counts);
    expert_ffn1<<<dim3(FDIM / BN, NTOK / BM, NGRP), 256, 0, stream>>>(
        x, w1, b1, sw1, sb1, lists, counts, hbuf);
    expert_ffn2<<<dim3(HDIM / BN, NTOK / BM, NGRP), 256, 0, stream>>>(
        hbuf, w2, b2, sw2, sb2, lists, counts, out);
}
